// SwinTransformerBlock_46548855554741
// MI455X (gfx1250) — matrix operand order repaired and checked by reading, not compiled, not run on hardware
//
#include <hip/hip_runtime.h>
#include <hip/hip_bf16.h>
#include <math.h>

// ---------- problem constants ----------
static constexpr int Bn = 32, Hn = 64, Wn = 64, Cn = 192;
static constexpr int HEADS = 6, WSZ = 8, SHIFT = 4;
static constexpr int HIDDEN = 4 * Cn;      // 768
static constexpr int NWIN = (Hn / WSZ) * (Wn / WSZ); // 64 windows / image
static constexpr int NTOK = WSZ * WSZ;     // 64 tokens / window
static constexpr int TOKENS = Bn * Hn * Wn;          // 131072
static constexpr int XS_LD = 200;          // padded LDS row stride (halves)
static constexpr int H1_LD = 264;          // padded LDS stride for MLP hidden chunk

typedef _Float16 h16;
typedef __attribute__((ext_vector_type(16))) _Float16 v16h;
typedef __attribute__((ext_vector_type(8)))  _Float16 v8h;
typedef __attribute__((ext_vector_type(8)))  float    v8f;
typedef __attribute__((ext_vector_type(4)))  int      v4i;

// ---------- async global->LDS staging (gfx1250 GLOBAL_LOAD_ASYNC_TO_LDS_B128) ----------
#if defined(__HIP_DEVICE_COMPILE__) && \
    __has_builtin(__builtin_amdgcn_global_load_async_to_lds_b128) && \
    __has_builtin(__builtin_amdgcn_s_wait_asynccnt)
#define HAVE_ASYNC 1
typedef __attribute__((address_space(1))) v4i av4i_g;  // global
typedef __attribute__((address_space(3))) v4i av4i_l;  // LDS
#else
#define HAVE_ASYNC 0
#endif

// copy one 16B chunk global->LDS (async if available)
__device__ __forceinline__ void cp16(h16* dst_lds, const h16* src_g) {
#if HAVE_ASYNC
  __builtin_amdgcn_global_load_async_to_lds_b128((av4i_g*)(v4i*)src_g,
                                                 (av4i_l*)(v4i*)dst_lds, 0, 0);
#else
  *(v4i*)dst_lds = *(const v4i*)src_g;
#endif
}
__device__ __forceinline__ void cp_drain() {
#if HAVE_ASYNC
  __builtin_amdgcn_s_wait_asynccnt(0);
#endif
}

// ---------- WMMA wrapper (v_wmma_f32_16x16x32_f16) ----------
__device__ __forceinline__ v8f wmma(v16h a, v16h b, v8f c) {
  return __builtin_amdgcn_wmma_f32_16x16x32_f16(false, a, false, b,
                                                (short)0, c, false, false);
}

// A operand: 16x32 tile from row-major [.., ld] f16 matrix.
// lanes 0-15 row M=lane, halves 0..15 = K 0..15;
// lanes 16-31 same rows, K 16..31.  (bases are 16B aligned by construction)
__device__ __forceinline__ v16h loadA_rm(const h16* p, int ld, int m0, int k0, int lane) {
  const h16* r = p + (m0 + (lane & 15)) * ld + k0 + ((lane >> 4) << 4);
  v8h lo = *(const v8h*)(r);
  v8h hi = *(const v8h*)(r + 8);
  v16h a;
#pragma unroll
  for (int i = 0; i < 8; ++i) { a[i] = lo[i]; a[8 + i] = hi[i]; }
  return a;
}

// B operand from pre-swizzled weights: tile (kt, ntg) stored lane-major,
// 16 contiguous halves per lane -> two global_load_b128 per operand.
__device__ __forceinline__ v16h loadB_sw(const h16* swz, int KT, int kt, int ntg, int lane) {
  const h16* r = swz + ((((size_t)ntg * KT + kt) * 32 + lane) << 4);
  v8h lo = *(const v8h*)(r);
  v8h hi = *(const v8h*)(r + 8);
  v16h b;
#pragma unroll
  for (int i = 0; i < 8; ++i) { b[i] = lo[i]; b[8 + i] = hi[i]; }
  return b;
}

// B operand: 32x16 tile, element B[k][n] = p[(k0+k)*ld + n0+n]  (row-major KxN, LDS)
__device__ __forceinline__ v16h loadB_rm(const h16* p, int ld, int k0, int n0, int lane) {
  const h16* r = p + (k0 + ((lane >> 4) << 4)) * ld + n0 + (lane & 15);
  v16h b;
#pragma unroll
  for (int i = 0; i < 16; ++i) b[i] = r[i * ld];
  return b;
}

// B operand as transpose of row-major [N][K] matrix: B[k][n] = p[(n0+n)*ld + k0+k]
// contiguous 16 halves (32B aligned for ld=32)
__device__ __forceinline__ v16h loadB_t(const h16* p, int ld, int k0, int n0, int lane) {
  const h16* r = p + (n0 + (lane & 15)) * ld + k0 + ((lane >> 4) << 4);
  v8h lo = *(const v8h*)(r);
  v8h hi = *(const v8h*)(r + 8);
  v16h b;
#pragma unroll
  for (int i = 0; i < 8; ++i) { b[i] = lo[i]; b[8 + i] = hi[i]; }
  return b;
}

// Store a 16x16 f32 D tile as f16 into row-major buffer: (d[j]+bias)*scale
__device__ __forceinline__ void storeD_f16(h16* p, int ld, int m0, int n0,
                                           v8f d, float bias, float scale, int lane) {
  int col = n0 + (lane & 15);
  int rb  = m0 + ((lane >> 4) << 3);
#pragma unroll
  for (int j = 0; j < 8; ++j)
    p[(rb + j) * ld + col] = (h16)((d[j] + bias) * scale);
}

__device__ __forceinline__ int lab(int u) { return (u < 56) ? 0 : (u < 60 ? 1 : 2); }

// ---------- kernel: fp32 weights -> f16, pre-swizzled into WMMA B-operand tile order ----
// dst[((ntg*KT + kt)*32 + lane)*16 + i] = W[kt*32 + ((lane>>4)<<4) + i][ntg*16 + (lane&15)]
__global__ void swz_kernel(const float* __restrict__ src, h16* __restrict__ dst,
                           int K, int N) {
  int gid = blockIdx.x * blockDim.x + threadIdx.x;
  int KT = K >> 5;
  int total = KT * (N >> 4) * 32;
  if (gid >= total) return;
  int lane = gid & 31;
  int tile = gid >> 5;
  int ntg = tile / KT, kt = tile % KT;
  int col  = ntg * 16 + (lane & 15);
  int krow = kt * 32 + ((lane >> 4) << 4);
  h16* d = dst + (size_t)gid * 16;
#pragma unroll
  for (int i = 0; i < 16; ++i) d[i] = (h16)src[(size_t)(krow + i) * N + col];
}

// ---------- kernel: LayerNorm (wave per token) -> f16 ----------
__global__ __launch_bounds__(256) void ln_kernel(const float* __restrict__ src,
                                                 const float* __restrict__ g,
                                                 const float* __restrict__ bta,
                                                 h16* __restrict__ dst) {
  int lane = threadIdx.x & 31;
  size_t tok = (size_t)blockIdx.x * 8 + (threadIdx.x >> 5);
  const float* p = src + tok * Cn;
  float v[6], s = 0.f;
#pragma unroll
  for (int i = 0; i < 6; ++i) { v[i] = p[lane + 32 * i]; s += v[i]; }
#pragma unroll
  for (int m = 1; m < 32; m <<= 1) s += __shfl_xor(s, m, 32);
  float mu = s * (1.f / 192.f), q = 0.f;
#pragma unroll
  for (int i = 0; i < 6; ++i) { float d = v[i] - mu; q += d * d; }
#pragma unroll
  for (int m = 1; m < 32; m <<= 1) q += __shfl_xor(q, m, 32);
  float rs = rsqrtf(q * (1.f / 192.f) + 1e-5f);
#pragma unroll
  for (int i = 0; i < 6; ++i) {
    int c = lane + 32 * i;
    dst[tok * Cn + c] = (h16)((v[i] - mu) * rs * g[c] + bta[c]);
  }
}

// ---------- kernel: windowed attention (per-window block, 2 waves x 3 heads) ----------
__global__ __launch_bounds__(64) void attn_kernel(const h16* __restrict__ X16,
                                                  const h16* __restrict__ Wqkv,  // swizzled, K=192,N=576
                                                  const float* __restrict__ bqkv,
                                                  const float* __restrict__ bias_table,
                                                  h16* __restrict__ AttnOut) {
  __shared__ h16 Xs[NTOK * XS_LD];   // 25600 B, shifted window tile (f16)
  __shared__ h16 SW[2][6144];        // per-wave: Q[0:2048) K[2048:4096) V[4096:6144); P over Q+K

  int blk = blockIdx.x;
  int b = blk >> 6, nw = blk & 63;
  int hw = nw >> 3, ww = nw & 7;
  int tid = threadIdx.x, lane = tid & 31, wv = tid >> 5;

  { // stage window with roll(-4,-4): row per thread, 12x 16B chunks (async to LDS)
    int r = tid >> 3, c = tid & 7;
    int hs = (hw * 8 + r + SHIFT) & 63;
    int wsx = (ww * 8 + c + SHIFT) & 63;
    const h16* srcp = X16 + ((size_t)((b * 64 + hs) * 64 + wsx)) * Cn;
    h16* dstp = Xs + tid * XS_LD;
#pragma unroll
    for (int i = 0; i < 12; ++i)
      cp16(dstp + i * 8, srcp + i * 8);
    cp_drain();
  }
  __syncthreads();

  h16* Qs = SW[wv];
  h16* Ks = SW[wv] + 2048;
  h16* Vs = SW[wv] + 4096;
  h16* Ps = SW[wv];            // 64x64, reuses Q+K after scores are in registers

  const float qscale = 0.17677669529663687f; // 32^-0.5
  int hiL = lane >> 4, cl = lane & 15;

  for (int hh = 0; hh < 3; ++hh) {
    int h = wv * 3 + hh;

    // ---- Q and K projections: [64x32] each, tiles 4(m) x 2(n), K=6x32 ----
#pragma unroll
    for (int sel = 0; sel < 2; ++sel) {
      h16* dstp = sel == 0 ? Qs : Ks;
      float sc = sel == 0 ? qscale : 1.0f;
#pragma unroll
      for (int mt = 0; mt < 4; ++mt)
#pragma unroll
        for (int nt = 0; nt < 2; ++nt) {
          v8f acc = {};
#pragma unroll
          for (int ks = 0; ks < 6; ++ks)
            acc = wmma(loadA_rm(Xs, XS_LD, mt * 16, ks * 32, lane),
                       loadB_sw(Wqkv, 6, ks, sel * 12 + h * 2 + nt, lane), acc);
          float bias = bqkv[sel * Cn + h * 32 + nt * 16 + cl];
          storeD_f16(dstp, 32, mt * 16, nt * 16, acc, bias, sc, lane);
        }
    }

    // ---- scores: Q[64x32] @ K^T[32x64] -> 16 tiles (f32 accum) ----
    v8f s[4][4];
#pragma unroll
    for (int mt = 0; mt < 4; ++mt)
#pragma unroll
      for (int nt = 0; nt < 4; ++nt) {
        v8f z = {};
        s[mt][nt] = wmma(loadA_rm(Qs, 32, mt * 16, 0, lane),
                         loadB_t(Ks, 32, 0, nt * 16, lane), z);
      }

    // ---- V projection (Xs still live) -> Vs ----
#pragma unroll
    for (int mt = 0; mt < 4; ++mt)
#pragma unroll
      for (int nt = 0; nt < 2; ++nt) {
        v8f acc = {};
#pragma unroll
        for (int ks = 0; ks < 6; ++ks)
          acc = wmma(loadA_rm(Xs, XS_LD, mt * 16, ks * 32, lane),
                     loadB_sw(Wqkv, 6, ks, 24 + h * 2 + nt, lane), acc);
        float bias = bqkv[2 * Cn + h * 32 + nt * 16 + cl];
        storeD_f16(Vs, 32, mt * 16, nt * 16, acc, bias, 1.0f, lane);
      }

    // ---- rel-pos bias + shift mask + row softmax (rows live in 16-lane groups) ----
#pragma unroll
    for (int mt = 0; mt < 4; ++mt) {
#pragma unroll
      for (int j = 0; j < 8; ++j) {
        int q = mt * 16 + 8 * hiL + j;
        int qr = q >> 3, qc = q & 7;
        int labq = lab(hw * 8 + qr) * 3 + lab(ww * 8 + qc);
        float vals[4], mx = -1e30f;
#pragma unroll
        for (int nt = 0; nt < 4; ++nt) {
          int key = nt * 16 + cl;
          int kr = key >> 3, kc = key & 7;
          int idx = (qr - kr + 7) * 15 + (qc - kc + 7);
          float bias = bias_table[idx * HEADS + h];
          int labk = lab(hw * 8 + kr) * 3 + lab(ww * 8 + kc);
          float val = s[mt][nt][j] + bias + ((labq != labk) ? -100.f : 0.f);
          vals[nt] = val; mx = fmaxf(mx, val);
        }
#pragma unroll
        for (int m = 1; m < 16; m <<= 1) mx = fmaxf(mx, __shfl_xor(mx, m, 32));
        float sum = 0.f;
#pragma unroll
        for (int nt = 0; nt < 4; ++nt) { vals[nt] = __expf(vals[nt] - mx); sum += vals[nt]; }
#pragma unroll
        for (int m = 1; m < 16; m <<= 1) sum += __shfl_xor(sum, m, 32);
        float inv = 1.f / sum;
#pragma unroll
        for (int nt = 0; nt < 4; ++nt) s[mt][nt][j] = vals[nt] * inv;
      }
    }

    // ---- store P (f16) to LDS for A-operand relayout ----
#pragma unroll
    for (int mt = 0; mt < 4; ++mt)
#pragma unroll
      for (int nt = 0; nt < 4; ++nt)
#pragma unroll
        for (int j = 0; j < 8; ++j)
          Ps[(mt * 16 + 8 * hiL + j) * 64 + nt * 16 + cl] = (h16)s[mt][nt][j];

    // ---- out_h = P[64x64] @ V[64x32], write to global at unrolled coords ----
#pragma unroll
    for (int mt = 0; mt < 4; ++mt)
#pragma unroll
      for (int ntv = 0; ntv < 2; ++ntv) {
        v8f o = {};
#pragma unroll
        for (int ks = 0; ks < 2; ++ks)
          o = wmma(loadA_rm(Ps, 64, mt * 16, ks * 32, lane),
                   loadB_rm(Vs, 32, ks * 32, ntv * 16, lane), o);
        int col = h * 32 + ntv * 16 + cl;
#pragma unroll
        for (int j = 0; j < 8; ++j) {
          int tkn = mt * 16 + 8 * hiL + j;
          int r = tkn >> 3, c = tkn & 7;
          int hf = (hw * 8 + r + SHIFT) & 63;
          int wf = (ww * 8 + c + SHIFT) & 63;
          AttnOut[((size_t)((b * 64 + hf) * 64 + wf)) * Cn + col] = (h16)o[j];
        }
      }
  }
}

// ---------- kernel: proj GEMM + bias + shortcut -> x1 (f32, d_out) ----------
__global__ __launch_bounds__(256) void proj_kernel(const h16* __restrict__ AttnOut,
                                                   const h16* __restrict__ Wp,   // swizzled K=192,N=192
                                                   const float* __restrict__ bp,
                                                   const float* __restrict__ xin,
                                                   float* __restrict__ x1out) {
  __shared__ h16 As[64 * XS_LD];
  size_t t0 = (size_t)blockIdx.x * 64;
  int tid = threadIdx.x, lane = tid & 31, wv = tid >> 5;
  for (int i = tid; i < 64 * 24; i += 256) {
    int row = i / 24, c = i % 24;
    cp16(As + row * XS_LD + c * 8, AttnOut + (t0 + row) * Cn + c * 8);
  }
  cp_drain();
  __syncthreads();
#pragma unroll
  for (int i = 0; i < 6; ++i) {
    int t = wv * 6 + i, mt = t / 12, nt = t % 12;
    v8f acc = {};
#pragma unroll
    for (int ks = 0; ks < 6; ++ks)
      acc = wmma(loadA_rm(As, XS_LD, mt * 16, ks * 32, lane),
                 loadB_sw(Wp, 6, ks, nt, lane), acc);
    int col = nt * 16 + (lane & 15);
    float bias = bp[col];
#pragma unroll
    for (int j = 0; j < 8; ++j) {
      size_t off = (t0 + mt * 16 + 8 * (lane >> 4) + j) * Cn + col;
      x1out[off] = acc[j] + bias + xin[off];
    }
  }
}

// ---------- kernel: fused MLP (fc1 + GELU + fc2) + residual, in-place on d_out ----------
__global__ __launch_bounds__(256) void mlp_kernel(const h16* __restrict__ Hln,
                                                  const h16* __restrict__ W1,   // swizzled K=192,N=768
                                                  const float* __restrict__ b1,
                                                  const h16* __restrict__ W2,   // swizzled K=768,N=192
                                                  const float* __restrict__ b2,
                                                  float* __restrict__ xout) {
  __shared__ h16 As[64 * XS_LD];   // 25600 B
  __shared__ h16 H1[64 * H1_LD];   // 33792 B (256-col chunk of hidden)
  size_t t0 = (size_t)blockIdx.x * 64;
  int tid = threadIdx.x, lane = tid & 31, wv = tid >> 5;
  for (int i = tid; i < 64 * 24; i += 256) {
    int row = i / 24, c = i % 24;
    cp16(As + row * XS_LD + c * 8, Hln + (t0 + row) * Cn + c * 8);
  }
  cp_drain();
  __syncthreads();

  v8f oacc[6];
#pragma unroll
  for (int i = 0; i < 6; ++i) oacc[i] = (v8f){};

  for (int ch = 0; ch < 3; ++ch) {
    // stream next chunk's fc1 weights toward L2 (global_prefetch_b8)
    if (ch < 2) {
      const h16* nxt = W1 + ((size_t)(ch + 1) * 16) * 6 * 512; // ntg block base
      __builtin_prefetch(nxt + (size_t)tid * 384, 0, 1);
    }
    __syncthreads();  // previous chunk's fc2 reads of H1 are done
    // fc1 chunk: [64x192] @ [192x256] -> GELU -> H1 (f16)
#pragma unroll
    for (int i = 0; i < 8; ++i) {
      int t = wv * 8 + i, mt = t >> 4, nt = t & 15;
      v8f acc = {};
#pragma unroll
      for (int ks = 0; ks < 6; ++ks)
        acc = wmma(loadA_rm(As, XS_LD, mt * 16, ks * 32, lane),
                   loadB_sw(W1, 6, ks, ch * 16 + nt, lane), acc);
      int col = ch * 256 + nt * 16 + (lane & 15);
      float bias = b1[col];
#pragma unroll
      for (int j = 0; j < 8; ++j) {
        float v = acc[j] + bias;
        v = 0.5f * v * (1.f + erff(v * 0.70710678118654752f));  // exact GELU
        H1[(mt * 16 + 8 * (lane >> 4) + j) * H1_LD + nt * 16 + (lane & 15)] = (h16)v;
      }
    }
    __syncthreads();
    // fc2 partial: accumulate [64x192] over this 256-wide K chunk
#pragma unroll
    for (int i = 0; i < 6; ++i) {
      int t = wv * 6 + i, mt = t / 12, nt = t % 12;
      v8f acc = oacc[i];
#pragma unroll
      for (int ks = 0; ks < 8; ++ks)
        acc = wmma(loadA_rm(H1, H1_LD, mt * 16, ks * 32, lane),
                   loadB_sw(W2, 24, ch * 8 + ks, nt, lane), acc);
      oacc[i] = acc;
    }
  }
  // residual add in place (block owns disjoint rows)
#pragma unroll
  for (int i = 0; i < 6; ++i) {
    int t = wv * 6 + i, mt = t / 12, nt = t % 12;
    int col = nt * 16 + (lane & 15);
    float bias = b2[col];
#pragma unroll
    for (int j = 0; j < 8; ++j) {
      size_t off = (t0 + mt * 16 + 8 * (lane >> 4) + j) * Cn + col;
      xout[off] = xout[off] + oacc[i][j] + bias;
    }
  }
}

// ---------- host launch ----------
extern "C" void kernel_launch(void* const* d_in, const int* in_sizes, int n_in,
                              void* d_out, int out_size, void* d_ws, size_t ws_size,
                              hipStream_t stream) {
  (void)in_sizes; (void)n_in; (void)out_size; (void)ws_size;
  const float* x      = (const float*)d_in[0];
  const float* ln1_g  = (const float*)d_in[1];
  const float* ln1_b  = (const float*)d_in[2];
  const float* qkv_w  = (const float*)d_in[3];
  const float* qkv_b  = (const float*)d_in[4];
  const float* proj_w = (const float*)d_in[5];
  const float* proj_b = (const float*)d_in[6];
  const float* btab   = (const float*)d_in[7];
  const float* ln2_g  = (const float*)d_in[8];
  const float* ln2_b  = (const float*)d_in[9];
  const float* fc1_w  = (const float*)d_in[10];
  const float* fc1_b  = (const float*)d_in[11];
  const float* fc2_w  = (const float*)d_in[12];
  const float* fc2_b  = (const float*)d_in[13];
  float* outp = (float*)d_out;

  const size_t tokC = (size_t)TOKENS * Cn;     // 25,165,824
  h16* X16    = (h16*)d_ws;                    // LN output / LN2 output (reused)
  h16* Attn16 = X16 + tokC;                    // attention output (pre-proj)
  h16* Wq16   = Attn16 + tokC;                 // swizzled weights
  h16* Wp16   = Wq16 + (size_t)Cn * 3 * Cn;    // 110592
  h16* W116   = Wp16 + (size_t)Cn * Cn;        // 36864
  h16* W216   = W116 + (size_t)Cn * HIDDEN;    // 147456

  // 1) weights -> f16, pre-swizzled into B-operand tile order
  {
    int t;
    t = (Cn / 32) * ((3 * Cn) / 16) * 32;   // qkv: 6*36*32
    swz_kernel<<<(t + 255) / 256, 256, 0, stream>>>(qkv_w, Wq16, Cn, 3 * Cn);
    t = (Cn / 32) * (Cn / 16) * 32;          // proj
    swz_kernel<<<(t + 255) / 256, 256, 0, stream>>>(proj_w, Wp16, Cn, Cn);
    t = (Cn / 32) * (HIDDEN / 16) * 32;      // fc1
    swz_kernel<<<(t + 255) / 256, 256, 0, stream>>>(fc1_w, W116, Cn, HIDDEN);
    t = (HIDDEN / 32) * (Cn / 16) * 32;      // fc2
    swz_kernel<<<(t + 255) / 256, 256, 0, stream>>>(fc2_w, W216, HIDDEN, Cn);
  }

  // 2) LN1 -> X16 (f16)
  ln_kernel<<<TOKENS / 8, 256, 0, stream>>>(x, ln1_g, ln1_b, X16);

  // 3) windowed attention (roll + partition + qkv + softmax + @V, unrolled coords)
  attn_kernel<<<Bn * NWIN, 64, 0, stream>>>(X16, Wq16, qkv_b, btab, Attn16);

  // 4) proj + shortcut -> x1 in d_out (f32)
  proj_kernel<<<TOKENS / 64, 256, 0, stream>>>(Attn16, Wp16, proj_b, x, outp);

  // 5) LN2 -> X16 (reused), then fused MLP + residual in place on d_out
  ln_kernel<<<TOKENS / 8, 256, 0, stream>>>(outp, ln2_g, ln2_b, X16);
  mlp_kernel<<<TOKENS / 64, 256, 0, stream>>>(X16, W116, fc1_b, W216, fc2_b, outp);
}
